// SelfLearningMFD_56255481643366
// MI455X (gfx1250) — matrix operand order repaired and checked by reading, not compiled, not run on hardware
//
#include <hip/hip_runtime.h>
#include <hip/hip_bf16.h>

typedef __attribute__((ext_vector_type(16))) __bf16 v16bf;
typedef __attribute__((ext_vector_type(8)))  __bf16 v8bf;
typedef __attribute__((ext_vector_type(8)))  float  v8f;

#define BATCH    8
#define C_IN     256
#define C_OUT    512
#define BEV      128
#define HW       (BEV * BEV)      // 16384
#define N_OBJ    128
#define OC_BLOCK 64
#define P_BLOCK  128
#define P_CHUNKS 2
#define MIN_OV   0.1f
#define N_GEMM_BLOCKS ((HW / (P_BLOCK * P_CHUNKS)) * (C_OUT / OC_BLOCK) * BATCH) // 64*8*8 = 4096

// ---------------------------------------------------------------------------
// Kernel A: per-batch foreground gaussian heatmap (max over 128 objects) +
// per-batch fg sum (deterministic in-block tree reduce).
// ---------------------------------------------------------------------------
__global__ __launch_bounds__(256)
void fg_kernel(const float* __restrict__ gt, float* __restrict__ fg_out,
               float* __restrict__ ws_fgsum) {
    const int b   = blockIdx.x;
    const int tid = threadIdx.x;

    __shared__ float o_valid[N_OBJ];
    __shared__ float o_r[N_OBJ];
    __shared__ float o_i2s2[N_OBJ];
    __shared__ float o_cx[N_OBJ];
    __shared__ float o_cy[N_OBJ];

    if (tid < N_OBJ) {
        const float* box = gt + ((size_t)b * N_OBJ + tid) * 9;
        const float w = box[3] * 1.25f;   // / VOXEL / OSF = /0.1/8
        const float l = box[4] * 1.25f;
        // gaussian_radius(height=l, width=w), min_overlap = 0.1
        const float b1 = l + w;
        const float c1 = w * l * (1.f - MIN_OV) / (1.f + MIN_OV);
        const float r1 = (b1 + sqrtf(fmaxf(b1 * b1 - 4.f * c1, 0.f))) * 0.5f;
        const float b2 = 2.f * (l + w);
        const float c2 = (1.f - MIN_OV) * w * l;
        const float r2 = (b2 + sqrtf(fmaxf(b2 * b2 - 16.f * c2, 0.f))) * 0.5f;
        const float a3 = 4.f * MIN_OV;
        const float b3 = -2.f * MIN_OV * (l + w);
        const float c3 = (MIN_OV - 1.f) * w * l;
        const float r3 = (b3 + sqrtf(fmaxf(b3 * b3 - 4.f * a3 * c3, 0.f))) * 0.5f;
        int r = (int)truncf(fminf(fminf(r1, r2), r3));
        if (r < 1) r = 1;
        const float cx = (box[0] + 51.2f) * 1.25f;
        const float cy = (box[1] + 51.2f) * 1.25f;
        const int cxi = (int)truncf(cx);
        const int cyi = (int)truncf(cy);
        const bool valid = (w > 0.f) && (l > 0.f) &&
                           (cxi >= 0) && (cxi < BEV) && (cyi >= 0) && (cyi < BEV);
        const float sigma = (2.f * (float)r + 1.f) / 6.f;
        o_valid[tid] = valid ? 1.f : 0.f;
        o_r[tid]     = (float)r;
        o_i2s2[tid]  = 1.f / (2.f * sigma * sigma);
        o_cx[tid]    = (float)cxi;
        o_cy[tid]    = (float)cyi;
    }
    __syncthreads();

    float lsum = 0.f;
    for (int i = 0; i < HW / 256; ++i) {
        const int p  = i * 256 + tid;
        const float px = (float)(p & 127);
        const float py = (float)(p >> 7);
        float best = 0.f;
        for (int n = 0; n < N_OBJ; ++n) {
            const float dx = px - o_cx[n];
            const float dy = py - o_cy[n];
            const float g  = __expf(-(dx * dx + dy * dy) * o_i2s2[n]);
            const bool inside = (fabsf(dx) <= o_r[n]) && (fabsf(dy) <= o_r[n]);
            const float v = (o_valid[n] != 0.f && inside) ? g : 0.f;
            best = fmaxf(best, v);
        }
        fg_out[(size_t)b * HW + p] = best;
        lsum += best;
    }

    __shared__ float red[256];
    red[tid] = lsum;
    __syncthreads();
    for (int s = 128; s > 0; s >>= 1) {
        if (tid < s) red[tid] += red[tid + s];
        __syncthreads();
    }
    if (tid == 0) ws_fgsum[b] = red[0];
}

// ---------------------------------------------------------------------------
// Kernel B: fused bf16-WMMA GEMM (W[512x256] @ X[256xHW]) + bias + weighted
// squared-error partial reduction.
// Block = 256 threads = 8 waves; block tile = 64 o x 128 p per chunk
// (2 chunks). Waves in a 2(o) x 4(p) grid; each wave register-blocks a
// 32x32 output tile (4 accumulators) so every K-step is
// 8 x ds_load_b128 -> 4 x v_wmma_f32_16x16x32_bf16 (2 LDS loads per WMMA).
// ---------------------------------------------------------------------------
__global__ __launch_bounds__(256)
void gemm_loss_kernel(const float* __restrict__ student,
                      const float* __restrict__ teacher,
                      const float* __restrict__ W,
                      const float* __restrict__ bias,
                      const float* __restrict__ fg,
                      float* __restrict__ partials) {
    const int pblk = blockIdx.x;   // 64 blocks of 256 pixels
    const int oblk = blockIdx.y;   // 8 blocks of 64 out-channels
    const int b    = blockIdx.z;   // batch
    const int tid  = threadIdx.x;

    // +8 bf16 (16B) padding keeps rows 16B-aligned and staggers banks.
    __shared__ __bf16 sW[OC_BLOCK][C_IN + 8];   // [o][c]
    __shared__ __bf16 sX[P_BLOCK][C_IN + 8];    // [p][c] (transposed on store)

    // Stage W slice once: 32B global reads, f32 -> bf16, one b128 LDS store
    // per 8 channels.
    const float* Wg = W + (size_t)(oblk * OC_BLOCK) * C_IN;
    for (int j = tid; j < (OC_BLOCK * C_IN) / 8; j += 256) {
        const int o = j >> 5;            // (j*8) / 256
        const int c = (j & 31) * 8;
        const float4 f0 = *(const float4*)(Wg + o * C_IN + c);
        const float4 f1 = *(const float4*)(Wg + o * C_IN + c + 4);
        v8bf pk;
        pk[0] = (__bf16)f0.x; pk[1] = (__bf16)f0.y;
        pk[2] = (__bf16)f0.z; pk[3] = (__bf16)f0.w;
        pk[4] = (__bf16)f1.x; pk[5] = (__bf16)f1.y;
        pk[6] = (__bf16)f1.z; pk[7] = (__bf16)f1.w;
        *(v8bf*)&sW[o][c] = pk;
    }

    const int lane = tid & 31;
    const int wave = tid >> 5;
    const int wo   = wave >> 2;            // 0..1 : o sub-tile (32 rows)
    const int wp   = wave & 3;             // 0..3 : p sub-tile (32 cols)
    const int lrow = lane & 15;
    const int lhi  = lane >> 4;

    const size_t sbase = (size_t)b * C_IN * HW + (size_t)pblk * (P_BLOCK * P_CHUNKS);
    const float* Tg = teacher + ((size_t)b * C_OUT + (size_t)oblk * OC_BLOCK) * HW;
    const float* Fg = fg + (size_t)b * HW;

    float lsum = 0.f;

    for (int chunk = 0; chunk < P_CHUNKS; ++chunk) {
        __syncthreads();   // previous chunk's WMMA fragment loads are done
        const int p0 = chunk * P_BLOCK;
        // Stage student tile 128p x 256c transposed into [p][c]:
        // 8 strided b32 global loads (lane-coalesced over p), pack to v8bf,
        // single ds_store_b128.
        for (int j = tid; j < (P_BLOCK * C_IN) / 8; j += 256) {
            const int p = j & (P_BLOCK - 1);
            const int c = (j >> 7) * 8;
            const float* src = student + sbase + (size_t)c * HW + p0 + p;
            v8bf pk;
#pragma unroll
            for (int e = 0; e < 8; ++e) pk[e] = (__bf16)src[(size_t)e * HW];
            *(v8bf*)&sX[p][c] = pk;
        }
        __syncthreads();

        // Prefetch next chunk's student rows into cache (global_prefetch_b8).
        if (chunk + 1 < P_CHUNKS) {
            __builtin_prefetch(student + sbase + (size_t)tid * HW + p0 + P_BLOCK, 0, 1);
        }

        v8f acc00 = {}, acc01 = {}, acc10 = {}, acc11 = {};
        const int arow0 = wo * 32 + lrow;
        const int arow1 = arow0 + 16;
        const int brow0 = wp * 32 + lrow;
        const int brow1 = brow0 + 16;

#pragma unroll
        for (int k = 0; k < C_IN / 32; ++k) {
            const int c0 = k * 32;
            union { v16bf v; v8bf h[2]; } A0, A1, B0, B1;
            // A (16x32 bf16): lanes<16 hold K=c0+0..7 / c0+16..23, lanes>=16 +8
            A0.h[0] = *(const v8bf*)&sW[arow0][c0 + lhi * 8];
            A0.h[1] = *(const v8bf*)&sW[arow0][c0 + 16 + lhi * 8];
            A1.h[0] = *(const v8bf*)&sW[arow1][c0 + lhi * 8];
            A1.h[1] = *(const v8bf*)&sW[arow1][c0 + 16 + lhi * 8];
            // B (32x16 bf16): lanes<16 hold K=c0+0..7 / c0+16..23, lanes>=16 +8
            B0.h[0] = *(const v8bf*)&sX[brow0][c0 + lhi * 8];
            B0.h[1] = *(const v8bf*)&sX[brow0][c0 + 16 + lhi * 8];
            B1.h[0] = *(const v8bf*)&sX[brow1][c0 + lhi * 8];
            B1.h[1] = *(const v8bf*)&sX[brow1][c0 + 16 + lhi * 8];
            acc00 = __builtin_amdgcn_wmma_f32_16x16x32_bf16(
                false, A0.v, false, B0.v, (short)0, acc00, false, false);
            acc01 = __builtin_amdgcn_wmma_f32_16x16x32_bf16(
                false, A0.v, false, B1.v, (short)0, acc01, false, false);
            acc10 = __builtin_amdgcn_wmma_f32_16x16x32_bf16(
                false, A1.v, false, B0.v, (short)0, acc10, false, false);
            acc11 = __builtin_amdgcn_wmma_f32_16x16x32_bf16(
                false, A1.v, false, B1.v, (short)0, acc11, false, false);
        }

        // Epilogue: D layout — lane gives N(pixel)=lrow, VGPR v gives M(o)=v+8*lhi.
        const int pg0 = pblk * (P_BLOCK * P_CHUNKS) + p0 + wp * 32 + lrow;
        const int pg1 = pg0 + 16;
        const float fg0 = Fg[pg0];
        const float fg1 = Fg[pg1];
#pragma unroll
        for (int v = 0; v < 8; ++v) {
            const int m = lhi * 8 + v;
            // o-tile 0 (rows wo*32 .. +15)
            {
                const int ol = wo * 32 + m;
                const float bo = bias[oblk * OC_BLOCK + ol];
                const float t0 = Tg[(size_t)ol * HW + pg0];
                const float t1 = Tg[(size_t)ol * HW + pg1];
                const float d0 = (acc00[v] + bo) - t0;
                const float d1 = (acc01[v] + bo) - t1;
                lsum += d0 * d0 * fg0 + d1 * d1 * fg1;
            }
            // o-tile 1 (rows wo*32+16 .. +31)
            {
                const int ol = wo * 32 + 16 + m;
                const float bo = bias[oblk * OC_BLOCK + ol];
                const float t0 = Tg[(size_t)ol * HW + pg0];
                const float t1 = Tg[(size_t)ol * HW + pg1];
                const float d0 = (acc10[v] + bo) - t0;
                const float d1 = (acc11[v] + bo) - t1;
                lsum += d0 * d0 * fg0 + d1 * d1 * fg1;
            }
        }
    }

    __shared__ float red[256];
    __syncthreads();
    red[tid] = lsum;
    __syncthreads();
    for (int s = 128; s > 0; s >>= 1) {
        if (tid < s) red[tid] += red[tid + s];
        __syncthreads();
    }
    if (tid == 0) {
        partials[(((size_t)b * 8) + oblk) * 64 + pblk] = red[0];
    }
}

// ---------------------------------------------------------------------------
// Kernel C: deterministic final reduction -> loss scalar.
// ---------------------------------------------------------------------------
__global__ __launch_bounds__(256)
void finalize_kernel(const float* __restrict__ ws, float* __restrict__ out) {
    const int tid = threadIdx.x;
    __shared__ float red[256];
    float s = 0.f;
    for (int i = tid; i < N_GEMM_BLOCKS; i += 256) s += ws[BATCH + i];
    red[tid] = s;
    __syncthreads();
    for (int k = 128; k > 0; k >>= 1) {
        if (tid < k) red[tid] += red[tid + k];
        __syncthreads();
    }
    if (tid == 0) {
        float den = 0.f;
        for (int b = 0; b < BATCH; ++b) den += ws[b];
        out[0] = (red[0] / den) * 0.01f;
    }
}

// ---------------------------------------------------------------------------
extern "C" void kernel_launch(void* const* d_in, const int* in_sizes, int n_in,
                              void* d_out, int out_size, void* d_ws, size_t ws_size,
                              hipStream_t stream) {
    const float* student = (const float*)d_in[0];  // (8,256,128,128)
    const float* teacher = (const float*)d_in[1];  // (8,512,128,128)
    const float* gt      = (const float*)d_in[2];  // (8,128,9)
    const float* W       = (const float*)d_in[3];  // (512,256)
    const float* bias    = (const float*)d_in[4];  // (512,)

    float* out    = (float*)d_out;     // [0]=loss, [1..]=fg (8*1*128*128)
    float* ws     = (float*)d_ws;      // [0..7]=fg sums, [8..8+4096)=loss partials
    float* fg_out = out + 1;

    fg_kernel<<<dim3(BATCH), dim3(256), 0, stream>>>(gt, fg_out, ws);
    gemm_loss_kernel<<<dim3(HW / (P_BLOCK * P_CHUNKS), C_OUT / OC_BLOCK, BATCH),
                      dim3(256), 0, stream>>>(
        student, teacher, W, bias, fg_out, ws + BATCH);
    finalize_kernel<<<dim3(1), dim3(256), 0, stream>>>(ws, out);
}
